// BertCounterFactLatentEditCompetitionTransformer_59648505806971
// MI455X (gfx1250) — matrix operand order repaired and checked by reading, not compiled, not run on hardware
//
#include <hip/hip_runtime.h>

#define BB 32
#define KK 5
#define LL 256
#define DD 1024
#define CH 4            // attention L-chunks per (b,k)
#define LCH (LL / CH)   // 64 rows per chunk
#define NEGV -9.0e15f
#define SCALE 0.03125f  // 1/sqrt(1024)

typedef __attribute__((ext_vector_type(16))) __bf16 v16bf;
typedef __attribute__((ext_vector_type(8)))  float  v8f;

__device__ __forceinline__ __bf16 f2bf(float f) {
  unsigned u = __builtin_bit_cast(unsigned, f);
  u += 0x7FFFu + ((u >> 16) & 1u);                 // round-to-nearest-even
  unsigned short h = (unsigned short)(u >> 16);
  return __builtin_bit_cast(__bf16, h);
}
__device__ __forceinline__ float bf2f(__bf16 b) {
  unsigned u = ((unsigned)__builtin_bit_cast(unsigned short, b)) << 16;
  return __builtin_bit_cast(float, u);
}

__device__ __forceinline__ float block_sum(float v, float* red) {
  int t = threadIdx.x;
  red[t] = v; __syncthreads();
  for (int off = 128; off > 0; off >>= 1) {
    if (t < off) red[t] += red[t + off];
    __syncthreads();
  }
  float r = red[0]; __syncthreads();
  return r;
}
__device__ __forceinline__ float block_max(float v, float* red) {
  int t = threadIdx.x;
  red[t] = v; __syncthreads();
  for (int off = 128; off > 0; off >>= 1) {
    if (t < off) red[t] = fmaxf(red[t], red[t + off]);
    __syncthreads();
  }
  float r = red[0]; __syncthreads();
  return r;
}

// ---------------------------------------------------------------------------
// Generic small-M GEMM: C = act(A(MxK) * W^T + bias), fp32 operands split into
// bf16 hi/lo, 3x v_wmma_f32_16x16x32_bf16 per 32-deep K step (fp32-accurate).
// BT=1: W is (N,K) row-major (PyTorch lin).  BT=0: W is (K,N) row-major.
// ACT: 0=none 1=tanh 2=relu. Block=256 threads=8 wave32 -> 16(M) x 128(N).
// ---------------------------------------------------------------------------
template<int BT, int ACT>
__global__ __launch_bounds__(256) void gemm_wmma(
    const float* __restrict__ A, const float* __restrict__ W,
    const float* __restrict__ bias, float* __restrict__ C,
    int M, int N, int Kd) {
  const int wave = threadIdx.x >> 5;
  const int lane = threadIdx.x & 31;
  const int h = lane >> 4;     // lane half
  const int r = lane & 15;     // A row / B col within tile
  const int m0 = blockIdx.y << 4;
  const int n0 = (blockIdx.x << 7) + (wave << 4);
  if (n0 >= N || m0 >= M) return;
  const int ma = m0 + r;
  const int nb = n0 + r;
  v8f acc = {0.f, 0.f, 0.f, 0.f, 0.f, 0.f, 0.f, 0.f};
  for (int k0 = 0; k0 < Kd; k0 += 32) {
    v16bf ahi, alo, bhi, blo;
    // A fragment: element e -> K = k0 + 16h + e
    const float* arow = A + (size_t)ma * Kd + k0 + (h << 4);
#pragma unroll
    for (int e = 0; e < 8; ++e) {
      const float x0 = arow[e];
      const float x1 = arow[e + 8];
      const __bf16 h0 = f2bf(x0), h1 = f2bf(x1);
      ahi[e] = h0;      alo[e] = f2bf(x0 - bf2f(h0));
      ahi[e + 8] = h1;  alo[e + 8] = f2bf(x1 - bf2f(h1));
    }
    // B fragment: element e -> K = k0 + 16h + e, column nb
#pragma unroll
    for (int e = 0; e < 16; ++e) {
      const int kk = k0 + (h << 4) + e;
      const float w = BT ? W[(size_t)nb * Kd + kk] : W[(size_t)kk * N + nb];
      const __bf16 wh = f2bf(w);
      bhi[e] = wh; blo[e] = f2bf(w - bf2f(wh));
    }
    acc = __builtin_amdgcn_wmma_f32_16x16x32_bf16(false, ahi, false, bhi, (short)0, acc, false, false);
    acc = __builtin_amdgcn_wmma_f32_16x16x32_bf16(false, ahi, false, blo, (short)0, acc, false, false);
    acc = __builtin_amdgcn_wmma_f32_16x16x32_bf16(false, alo, false, bhi, (short)0, acc, false, false);
  }
  const int nc = n0 + r;
  const float bv = bias ? bias[nc] : 0.f;
#pragma unroll
  for (int v = 0; v < 8; ++v) {
    const int mm = m0 + (h << 3) + v;   // C layout: m = 8h + v, n = lane&15
    float val = acc[v] + bv;
    if (ACT == 1) val = tanhf(val);
    if (ACT == 2) val = fmaxf(val, 0.f);
    C[(size_t)mm * N + nc] = val;
  }
}

// ----------------- masks: sep_pos / vlen per (b,k) -------------------------
__global__ __launch_bounds__(256) void k_masks(const int* __restrict__ ids,
    const int* __restrict__ pad_idx, const int* __restrict__ sep_idx,
    int* __restrict__ sep_pos, int* __restrict__ vlen) {
  __shared__ int sv[256]; __shared__ int ss[256];
  const int g = blockIdx.x, l = threadIdx.x;
  const int pad = *pad_idx, sep = *sep_idx;
  const int id = ids[(size_t)g * LL + l];
  sv[l] = (id != pad) ? 1 : 0;
  ss[l] = (id == sep) ? l : LL;
  __syncthreads();
  for (int off = 128; off > 0; off >>= 1) {
    if (l < off) { sv[l] += sv[l + off]; ss[l] = ss[l] < ss[l + off] ? ss[l] : ss[l + off]; }
    __syncthreads();
  }
  if (l == 0) {
    const int vl = sv[0];
    int sp;
    if (ss[0] < LL) sp = ss[0];
    else { int fb = vl >> 1; fb = fb < 1 ? 1 : fb; fb = fb > (LL - 2) ? (LL - 2) : fb; sp = fb; }
    sep_pos[g] = sp; vlen[g] = vl;
  }
}

// ------------- sft = mean_k x ; al = sft . a_w + a_b -----------------------
__global__ __launch_bounds__(256) void k_sft_al(const float* __restrict__ x,
    const float* __restrict__ a_w, const float* __restrict__ a_b,
    float* __restrict__ sft, float* __restrict__ al) {
  __shared__ float red[256];
  const int l = blockIdx.x, b = blockIdx.y, t = threadIdx.x;
  const int d = t << 2;
  const float* base = x + ((size_t)b * KK * LL + l) * DD + d;
  float4 s = make_float4(0.f, 0.f, 0.f, 0.f);
#pragma unroll
  for (int k = 0; k < KK; ++k) {
    __builtin_prefetch(base + (size_t)k * LL * DD + DD, 0, 1);
    const float4 v = *(const float4*)(base + (size_t)k * LL * DD);
    s.x += v.x; s.y += v.y; s.z += v.z; s.w += v.w;
  }
  s.x *= 0.2f; s.y *= 0.2f; s.z *= 0.2f; s.w *= 0.2f;
  *(float4*)(sft + ((size_t)b * LL + l) * DD + d) = s;
  const float4 aw = *(const float4*)(a_w + d);
  float p = s.x * aw.x + s.y * aw.y + s.z * aw.z + s.w * aw.w;
  p = block_sum(p, red);
  if (t == 0) al[b * LL + l] = p + a_b[0];
}

// ------- gate softmax over L, sfs = sum gate*sft, sfc = masked mean --------
// grid (DD/256, BB): each block recomputes the (cheap) gate and reduces one
// 256-column slice of sft over L.
__global__ __launch_bounds__(256) void k_gate(const int* __restrict__ ids,
    const int* __restrict__ pad_idx,
    const int* __restrict__ sep_pos,
    const float* __restrict__ al, const float* __restrict__ sft,
    float* __restrict__ sfs, float* __restrict__ sfc) {
  __shared__ float red[256];
  __shared__ float gate[256];
  __shared__ float mskf[256];
  const int b = blockIdx.y, t = threadIdx.x;
  const int c = (blockIdx.x << 8) + t;   // column owned by this thread
  const int g0 = b * KK;                 // shared false mask comes from k=0
  const int sp = sep_pos[g0];
  const int pad = *pad_idx;
  const int id = ids[(size_t)g0 * LL + t];
  const int msk = (t > 0) && (t < sp) && (id != pad);
  const float a = al[b * LL + t];
  const float ml = msk ? a : NEGV;
  const float mx = block_max(ml, red);
  const float e = __expf(ml - mx);
  const float em = msk ? e : 0.f;
  const float s = block_sum(em, red);
  gate[t] = em / fmaxf(s, 1e-8f);
  mskf[t] = msk ? 1.f : 0.f;
  const float cnt = block_sum(mskf[t], red);
  const float inv_cnt = 1.f / fmaxf(cnt, 1.f);
  float as = 0.f, ac = 0.f;
  const float* sb = sft + (size_t)b * LL * DD + c;
  for (int ll = 0; ll < LL; ++ll) {
    const float v = sb[(size_t)ll * DD];
    as += gate[ll] * v;
    ac += mskf[ll] * v;
  }
  sfs[(size_t)b * DD + c] = as;
  sfc[(size_t)b * DD + c] = ac * inv_cnt;
}

// --------------------- row dot a fixed vector ------------------------------
__global__ __launch_bounds__(256) void k_dot_rows(const float* __restrict__ Ar,
    const float* __restrict__ vec, float* __restrict__ out, int ncols) {
  __shared__ float red[256];
  const int b = blockIdx.x, t = threadIdx.x;
  float p = 0.f;
  for (int d = t; d < ncols; d += 256) p += Ar[(size_t)b * ncols + d] * vec[d];
  p = block_sum(p, red);
  if (t == 0) out[b] = p;
}

__global__ __launch_bounds__(256) void k_rownorm2(const float* __restrict__ A,
                                                  float* __restrict__ out) {
  __shared__ float red[256];
  const int b = blockIdx.x, t = threadIdx.x;
  float p = 0.f;
  for (int d = t; d < DD; d += 256) { const float v = A[(size_t)b * DD + d]; p += v * v; }
  p = block_sum(p, red);
  if (t == 0) out[b] = p;
}

__global__ __launch_bounds__(256) void k_s2(const float* __restrict__ sfs,
    const float* __restrict__ delta, float* __restrict__ S2) {
  const int b = blockIdx.x, t = threadIdx.x;
  for (int d = t; d < DD; d += 256) {
    const float sv = sfs[(size_t)b * DD + d];
    S2[(size_t)b * DD + d] = sv;
    S2[(size_t)(BB + b) * DD + d] = sv + delta[(size_t)b * DD + d];
  }
}

// ---------------------------------------------------------------------------
// Single-pass (online-softmax) attention partials.  One block per (b,k,chunk),
// 64 rows per chunk processed 8-at-a-time:  each wave computes one row's logit
// while staging the row into LDS; then all 256 threads rescale/accumulate the
// running (m, s, acc) summary from LDS.  x is read exactly once.
// ---------------------------------------------------------------------------
__global__ __launch_bounds__(256) void k_attn_part(const float* __restrict__ x,
    const int* __restrict__ ids, const int* __restrict__ pad_idx,
    const int* __restrict__ sep_pos, const int* __restrict__ vlen,
    const float* __restrict__ V, const float* __restrict__ cvec,
    float* __restrict__ pacc, float* __restrict__ pms) {
  __shared__ float vsh[DD];        // 4 KB
  __shared__ float rows[8 * DD];   // 32 KB staging
  __shared__ float lgbuf[8];
  const int part = blockIdx.x;
  const int g = part / CH, chk = part % CH;
  const int b = g / KK;
  const int t = threadIdx.x;
  const int w = t >> 5, c = t & 31;
  const float* xb = x + (size_t)g * LL * DD;
  for (int i = t; i < DD; i += 256) vsh[i] = V[(size_t)b * DD + i];
  const int sp = sep_pos[g];
  int eosr = vlen[g] - 1; if (eosr < 0) eosr = 0;
  const int pad = *pad_idx;
  const float cv = cvec[b];
  __syncthreads();

  const int d0 = t << 2;
  float m = -3.0e38f, s = 0.f;
  float4 acc = make_float4(0.f, 0.f, 0.f, 0.f);
  const int lbeg = chk * LCH, lend = lbeg + LCH;
  for (int l0 = lbeg; l0 < lend; l0 += 8) {
    const int l = l0 + w;                       // row handled by this wave
    const float* row = xb + (size_t)l * DD;
    if (l + 8 < lend) __builtin_prefetch(row + 8 * DD + (c << 2), 0, 1);
    float p = 0.f;
#pragma unroll
    for (int j = 0; j < DD / 128; ++j) {        // 32 lanes x float4 = 128/iter
      const int dd = (c << 2) + (j << 7);
      const float4 v = *(const float4*)(row + dd);
      *(float4*)(rows + w * DD + dd) = v;
      const float4 q = *(const float4*)(vsh + dd);
      p += v.x * q.x + v.y * q.y + v.z * q.z + v.w * q.w;
    }
#pragma unroll
    for (int off = 16; off > 0; off >>= 1) p += __shfl_down(p, off, 32);
    if (c == 0) {
      const int id = ids[(size_t)g * LL + l];
      const int om = (l > sp) && (l < eosr) && (id != pad);
      const float lg = (p + cv) * SCALE;
      lgbuf[w] = om ? lg : NEGV;
    }
    __syncthreads();
    float lgs[8], gm = NEGV;
#pragma unroll
    for (int j = 0; j < 8; ++j) { lgs[j] = lgbuf[j]; gm = fmaxf(gm, lgs[j]); }
    const float nm = fmaxf(m, gm);
    const float sc = __expf(m - nm);
    acc.x *= sc; acc.y *= sc; acc.z *= sc; acc.w *= sc;
    s *= sc;
#pragma unroll
    for (int j = 0; j < 8; ++j) {
      const float wj = __expf(lgs[j] - nm);
      s += wj;
      const float4 rv = *(const float4*)(rows + j * DD + d0);
      acc.x += wj * rv.x; acc.y += wj * rv.y; acc.z += wj * rv.z; acc.w += wj * rv.w;
    }
    m = nm;
    __syncthreads();   // rows reused next iteration
  }
  *(float4*)(pacc + (size_t)part * DD + d0) = acc;
  if (t == 0) { pms[part * 2] = m; pms[part * 2 + 1] = s; }
}

// Merge CH chunk partials:  es = sum_ch e^(m_ch-M) acc_ch / sum_ch e^(m_ch-M) s_ch
__global__ __launch_bounds__(256) void k_attn_merge(const float* __restrict__ pacc,
    const float* __restrict__ pms, float* __restrict__ es) {
  const int g = blockIdx.x, t = threadIdx.x;
  float mch[CH], sch[CH];
  float M = -3.0e38f;
#pragma unroll
  for (int ch = 0; ch < CH; ++ch) {
    mch[ch] = pms[(g * CH + ch) * 2];
    sch[ch] = pms[(g * CH + ch) * 2 + 1];
    M = fmaxf(M, mch[ch]);
  }
  float wch[CH], stot = 0.f;
#pragma unroll
  for (int ch = 0; ch < CH; ++ch) { wch[ch] = __expf(mch[ch] - M); stot += wch[ch] * sch[ch]; }
  const float inv = 1.f / stot;
  const int d0 = t << 2;
  float4 o = make_float4(0.f, 0.f, 0.f, 0.f);
#pragma unroll
  for (int ch = 0; ch < CH; ++ch) {
    const float4 a = *(const float4*)(pacc + (size_t)(g * CH + ch) * DD + d0);
    const float wc = wch[ch];
    o.x += wc * a.x; o.y += wc * a.y; o.z += wc * a.z; o.w += wc * a.w;
  }
  o.x *= inv; o.y *= inv; o.z *= inv; o.w *= inv;
  *(float4*)(es + (size_t)g * DD + d0) = o;
}

// ---------- build MLP inputs: feats (4D) and evin (3D) ---------------------
__global__ __launch_bounds__(256) void k_feats(const float* __restrict__ sfs,
    const float* __restrict__ sfc, const float* __restrict__ es,
    float* __restrict__ feats, float* __restrict__ evin) {
  const int g = blockIdx.x, b = g / KK, t = threadIdx.x;
  float* F = feats + (size_t)g * 4 * DD;
  float* E = evin + (size_t)g * 3 * DD;
  for (int d = t; d < DD; d += 256) {
    const float fc = sfc[(size_t)b * DD + d];
    const float fs = sfs[(size_t)b * DD + d];
    const float ev = es[(size_t)g * DD + d];
    F[d] = fc; F[DD + d] = ev; F[2 * DD + d] = fabsf(fc - ev); F[3 * DD + d] = fc * ev;
    E[d] = fs; E[DD + d] = ev; E[2 * DD + d] = fs * ev;
  }
}

// ---------- option scores: eg + 0.15*repair + 0.05*min_pen - mean ----------
__global__ __launch_bounds__(256) void k_scores(const float* __restrict__ H1,
    const float* __restrict__ es2_w, const float* __restrict__ es2_b,
    const float* __restrict__ FQED, const float* __restrict__ EK,
    const float* __restrict__ dn, float* __restrict__ scores) {
  __shared__ float red[256];
  __shared__ float raw[KK];
  const int b = blockIdx.x, t = threadIdx.x;
  const float mp = -0.1f * dn[b];   // min_pen = -EDIT_MIN_W * |delta|^2
  for (int k = 0; k < KK; ++k) {
    const int g = b * KK + k;
    float p1 = 0.f, p2 = 0.f, p3 = 0.f;
    for (int d = t; d < DD; d += 256) {
      const float ek = EK[(size_t)g * DD + d];
      p1 += es2_w[d] * H1[(size_t)g * DD + d];
      p2 += FQED[(size_t)b * DD + d] * ek;          // fq . ek
      p3 += FQED[(size_t)(BB + b) * DD + d] * ek;   // edq . ek
    }
    const float s1 = block_sum(p1, red);
    const float s2 = block_sum(p2, red);
    const float s3 = block_sum(p3, red);
    if (t == 0) raw[k] = (s1 + es2_b[0]) + 0.15f * (s3 - s2) * SCALE + 0.05f * mp;
    __syncthreads();
  }
  if (t == 0) {
    float m = 0.f;
    for (int k = 0; k < KK; ++k) m += raw[k];
    m *= (1.f / KK);
    for (int k = 0; k < KK; ++k) scores[b * KK + k] = raw[k] - m;
  }
}

// ------------------------------ layernorm ----------------------------------
__global__ __launch_bounds__(256) void k_ln(const float* __restrict__ EV,
    const float* __restrict__ gam, const float* __restrict__ bet,
    float* __restrict__ outv) {
  __shared__ float red[256];
  const int g = blockIdx.x, t = threadIdx.x;
  const float* row = EV + (size_t)g * DD;
  float s = 0.f;
  for (int d = t; d < DD; d += 256) s += row[d];
  const float mu = block_sum(s, red) * (1.f / DD);
  float vs = 0.f;
  for (int d = t; d < DD; d += 256) { const float z = row[d] - mu; vs += z * z; }
  const float var = block_sum(vs, red) * (1.f / DD);
  const float inv = rsqrtf(var + 1e-5f);
  for (int d = t; d < DD; d += 256)
    outv[(size_t)g * DD + d] = (row[d] - mu) * inv * gam[d] + bet[d];
}

extern "C" void kernel_launch(void* const* d_in, const int* in_sizes, int n_in,
                              void* d_out, int out_size, void* d_ws, size_t ws_size,
                              hipStream_t stream) {
  (void)in_sizes; (void)n_in; (void)out_size; (void)ws_size;
  const float* x     = (const float*)d_in[0];
  const int*   ids   = (const int*)d_in[1];
  const int*   padp  = (const int*)d_in[2];
  const int*   sepp  = (const int*)d_in[3];
  const float* a_w   = (const float*)d_in[4];
  const float* a_b   = (const float*)d_in[5];
  const float* oq_w  = (const float*)d_in[6];
  const float* oq_b  = (const float*)d_in[7];
  const float* ok_w  = (const float*)d_in[8];
  const float* ok_b  = (const float*)d_in[9];
  const float* d_w   = (const float*)d_in[10];
  const float* d_b   = (const float*)d_in[11];
  const float* eq_w  = (const float*)d_in[12];
  const float* eq_b  = (const float*)d_in[13];
  const float* ek_w  = (const float*)d_in[14];
  const float* ek_b  = (const float*)d_in[15];
  const float* es1_w = (const float*)d_in[16];
  const float* es1_b = (const float*)d_in[17];
  const float* es2_w = (const float*)d_in[18];
  const float* es2_b = (const float*)d_in[19];
  const float* f1_w  = (const float*)d_in[20];
  const float* f1_b  = (const float*)d_in[21];
  const float* f2_w  = (const float*)d_in[22];
  const float* f2_b  = (const float*)d_in[23];
  const float* ln_g  = (const float*)d_in[24];
  const float* ln_b  = (const float*)d_in[25];

  float* ws = (float*)d_ws;
  size_t off = 0;
  auto alloc = [&](size_t n) { float* p = ws + off; off += n; return p; };
  float* sft   = alloc((size_t)BB * LL * DD);
  float* al    = alloc((size_t)BB * LL);
  float* sfs   = alloc((size_t)BB * DD);
  float* sfc   = alloc((size_t)BB * DD);
  float* OQ    = alloc((size_t)BB * DD);
  float* Vv    = alloc((size_t)BB * DD);
  float* delta = alloc((size_t)BB * DD);
  float* S2    = alloc((size_t)2 * BB * DD);
  float* FQED  = alloc((size_t)2 * BB * DD);
  float* ES    = alloc((size_t)BB * KK * DD);
  float* EK    = alloc((size_t)BB * KK * DD);
  float* FEATS = alloc((size_t)BB * KK * 4 * DD);
  float* H1    = alloc((size_t)BB * KK * DD);
  float* EVIN  = alloc((size_t)BB * KK * 3 * DD);
  float* H2    = alloc((size_t)BB * KK * DD);
  float* EVm   = alloc((size_t)BB * KK * DD);
  float* pacc  = alloc((size_t)BB * KK * CH * DD);
  float* pms   = alloc((size_t)BB * KK * CH * 2);
  float* cvec  = alloc(BB);
  float* dn    = alloc(BB);
  int* sep_pos = (int*)(ws + off); off += BB * KK;
  int* vlenp   = (int*)(ws + off); off += BB * KK;

  k_masks <<<BB * KK, 256, 0, stream>>>(ids, padp, sepp, sep_pos, vlenp);
  k_sft_al<<<dim3(LL, BB), 256, 0, stream>>>(x, a_w, a_b, sft, al);
  k_gate  <<<dim3(DD / 256, BB), 256, 0, stream>>>(ids, padp, sep_pos, al, sft, sfs, sfc);

  // OQ = sfs @ oq_w^T + oq_b ;  V = OQ @ ok_w ;  cvec = OQ . ok_b
  gemm_wmma<1, 0><<<dim3(DD / 128, BB / 16), 256, 0, stream>>>(sfs, oq_w, oq_b, OQ, BB, DD, DD);
  k_dot_rows<<<BB, 256, 0, stream>>>(OQ, ok_b, cvec, DD);
  gemm_wmma<0, 0><<<dim3(DD / 128, BB / 16), 256, 0, stream>>>(OQ, ok_w, nullptr, Vv, BB, DD, DD);

  // shared_delta = tanh(sfs @ d_w^T + d_b); |delta|^2; stacked eq inputs
  gemm_wmma<1, 1><<<dim3(DD / 128, BB / 16), 256, 0, stream>>>(sfs, d_w, d_b, delta, BB, DD, DD);
  k_rownorm2<<<BB, 256, 0, stream>>>(delta, dn);
  k_s2<<<BB, 256, 0, stream>>>(sfs, delta, S2);
  gemm_wmma<1, 0><<<dim3(DD / 128, (2 * BB) / 16), 256, 0, stream>>>(S2, eq_w, eq_b, FQED, 2 * BB, DD, DD);

  // single-pass online-softmax attention over x (x read once), then merge
  k_attn_part <<<BB * KK * CH, 256, 0, stream>>>(x, ids, padp, sep_pos, vlenp, Vv, cvec, pacc, pms);
  k_attn_merge<<<BB * KK, 256, 0, stream>>>(pacc, pms, ES);
  k_feats<<<BB * KK, 256, 0, stream>>>(sfs, sfc, ES, FEATS, EVIN);

  gemm_wmma<1, 0><<<dim3(DD / 128, (BB * KK) / 16), 256, 0, stream>>>(ES,    ek_w,  ek_b,  EK,  BB * KK, DD, DD);
  gemm_wmma<1, 2><<<dim3(DD / 128, (BB * KK) / 16), 256, 0, stream>>>(FEATS, es1_w, es1_b, H1,  BB * KK, DD, 4 * DD);
  gemm_wmma<1, 2><<<dim3(DD / 128, (BB * KK) / 16), 256, 0, stream>>>(EVIN,  f1_w,  f1_b,  H2,  BB * KK, DD, 3 * DD);
  gemm_wmma<1, 0><<<dim3(DD / 128, (BB * KK) / 16), 256, 0, stream>>>(H2,    f2_w,  f2_b,  EVm, BB * KK, DD, DD);

  float* outf = (float*)d_out;
  k_ln    <<<BB * KK, 256, 0, stream>>>(EVm, ln_g, ln_b, outf);
  k_scores<<<BB, 256, 0, stream>>>(H1, es2_w, es2_b, FQED, EK, dn, outf + (size_t)BB * KK * DD);
}
